// Rank2DecompositionEdgeBlock_15006615734322
// MI455X (gfx1250) — matrix operand order repaired and checked by reading, not compiled, not run on hardware
//
#include <hip/hip_runtime.h>
#include <hip/hip_bf16.h>
#include <math.h>

#define EMB 128

typedef __attribute__((ext_vector_type(16))) __bf16         v16bf;
typedef __attribute__((ext_vector_type(8)))  float          v8f;
typedef __attribute__((ext_vector_type(16))) unsigned short v16u;

#define NEG_LOG2E (-1.4426950408889634f)
#define NEG_LN2   (-0.6931471805599453f)

static __device__ __forceinline__ unsigned short f2bf(float f) {
    union { float f; unsigned int u; } v; v.f = f;
    unsigned int r = v.u + 0x7FFFu + ((v.u >> 16) & 1u); // round-to-nearest-even
    return (unsigned short)(r >> 16);
}

// Given t = -log2e * x, returns silu(x) / (-ln2) = t * rcp(1 + 2^t).
// Caller folds the (-ln2) into the output weight.
static __device__ __forceinline__ float silu_t(float t) {
    const float e = __builtin_amdgcn_exp2f(t);
    return t * __builtin_amdgcn_rcpf(1.0f + e);
}

// ---------------------------------------------------------------------------
// Main fused kernel: per 16-edge tile, GEMM [16,128]x[128,256] in bf16 WMMA
// (columns 0..127 = Ws1^T, 128..255 = Wi1^T), fused bias+silu+dot-with-W2,
// then segmented-sum into per-node accumulators with atomics.
// ---------------------------------------------------------------------------
__global__ __launch_bounds__(256) void edge_kernel(
    const float* __restrict__ x_edge, const float* __restrict__ edge_vec,
    const int* __restrict__ edge_index,
    const float* __restrict__ Ws1, const float* __restrict__ bs1,
    const float* __restrict__ Ws2, const float* __restrict__ bs2,
    const float* __restrict__ Wi1, const float* __restrict__ bi1,
    const float* __restrict__ Wi2, const float* __restrict__ bi2,
    float* __restrict__ nodeScalar, float* __restrict__ nodeIrrep,
    float* __restrict__ nodeCnt, int E)
{
    __shared__ unsigned short Wlds[256 * EMB]; // 64 KB: rows 0..127 Ws1, 128..255 Wi1 (bf16)
    __shared__ float auxW2[256];               // -ln2 * {Ws2 | Wi2}
    __shared__ float auxB1[256];               // -log2e * {bs1 | bi1}

    const int tid = threadIdx.x;
    for (int i = tid; i < 256 * EMB; i += blockDim.x) {
        const int r = i >> 7, k = i & 127;
        const float w = (r < 128) ? Ws1[r * EMB + k] : Wi1[(r - 128) * EMB + k];
        Wlds[i] = f2bf(w);
    }
    if (tid < 256) {
        auxW2[tid] = NEG_LN2 * ((tid < 128) ? Ws2[tid] : Wi2[tid - 128]);
        auxB1[tid] = NEG_LOG2E * ((tid < 128) ? bs1[tid] : bi1[tid - 128]);
    }
    __syncthreads();

    const int lane = tid & 31;
    const int n16  = lane & 15;   // A: row M / B: col N / C: col N
    const int g    = lane >> 4;   // lane-group (K-half select)

    const int wavesPerBlock = blockDim.x >> 5;
    const int wid      = blockIdx.x * wavesPerBlock + (tid >> 5);
    const int wstride  = gridDim.x * wavesPerBlock;
    const int numTiles = (E + 15) >> 4;

    const float SQRT3   = 1.7320508075688772f;
    const float SH2_INT = 0.63078313050504f; // sqrt(5/(4*pi))
    const float c_bs2 = bs2[0];
    const float c_bi2 = bi2[0];

    for (int tile = wid; tile < numTiles; tile += wstride) {
        const int base = tile << 4;

        // ---- A fragments (16x32 bf16, contiguous K): lane=M, halves split by g
        v16u afr[4];
        {
            int erow = base + n16; if (erow >= E) erow = E - 1;
            const float* xr = x_edge + (size_t)erow * EMB;
            #pragma unroll
            for (int kk = 0; kk < 4; ++kk) {
                const int c0 = 32 * kk + 16 * g;  // halves 0..7  -> K = c0..c0+7
                const int c1 = c0 + 8;            // halves 8..15 -> K = c1..c1+7
                float4 a0 = *(const float4*)(xr + c0);
                float4 a1 = *(const float4*)(xr + c0 + 4);
                float4 b0 = *(const float4*)(xr + c1);
                float4 b1 = *(const float4*)(xr + c1 + 4);
                v16u u;
                u[0]  = f2bf(a0.x); u[1]  = f2bf(a0.y); u[2]  = f2bf(a0.z); u[3]  = f2bf(a0.w);
                u[4]  = f2bf(a1.x); u[5]  = f2bf(a1.y); u[6]  = f2bf(a1.z); u[7]  = f2bf(a1.w);
                u[8]  = f2bf(b0.x); u[9]  = f2bf(b0.y); u[10] = f2bf(b0.z); u[11] = f2bf(b0.w);
                u[12] = f2bf(b1.x); u[13] = f2bf(b1.y); u[14] = f2bf(b1.z); u[15] = f2bf(b1.w);
                afr[kk] = u;
            }
        }

        // ---- pre-scaled spherical harmonics (-log2e * sph) for this lane-group
        float sphs[8][5];
        #pragma unroll
        for (int j = 0; j < 8; ++j) {
            int e = base + j + 8 * g; if (e >= E) e = E - 1;
            const float vx = edge_vec[3 * e + 0];
            const float vy = edge_vec[3 * e + 1];
            const float vz = edge_vec[3 * e + 2];
            const float inv = rsqrtf(vx * vx + vy * vy + vz * vz);
            const float X = vx * inv, Y = vy * inv, Z = vz * inv;
            const float s = NEG_LOG2E * 0.63078313050504f; // fold SH2_INT too
            sphs[j][0] = s * (SQRT3 * X * Z);
            sphs[j][1] = s * (SQRT3 * X * Y);
            sphs[j][2] = s * (Y * Y - 0.5f * (X * X + Z * Z));
            sphs[j][3] = s * (SQRT3 * Y * Z);
            sphs[j][4] = s * (0.5f * SQRT3 * (Z * Z - X * X));
        }
        (void)SH2_INT;

        float sacc[8];
        float iacc[8][5];
        #pragma unroll
        for (int j = 0; j < 8; ++j) {
            sacc[j] = 0.0f;
            #pragma unroll
            for (int m = 0; m < 5; ++m) iacc[j][m] = 0.0f;
        }

        // ---- 16 N-tiles of the fused [128 x 256] weight matrix
        for (int t = 0; t < 16; ++t) {
            v8f acc = {0.f, 0.f, 0.f, 0.f, 0.f, 0.f, 0.f, 0.f};
            const int r = 16 * t + n16; // output channel handled by this lane
            #pragma unroll
            for (int kk = 0; kk < 4; ++kk) {
                // B fragment (32x16 bf16): lane=N, halves = 16 contiguous K
                const v16u bu = *(const v16u*)&Wlds[r * EMB + 32 * kk + 16 * g];
                acc = __builtin_amdgcn_wmma_f32_16x16x32_bf16(
                        false, __builtin_bit_cast(v16bf, afr[kk]),
                        false, __builtin_bit_cast(v16bf, bu),
                        (short)0, acc, false, false);
            }
            // C layout: acc[j], lane l -> edge m = j + 8*(l>>4), channel 16t + (l&15)
            const float w2 = auxW2[r]; // already * -ln2
            const float b1 = auxB1[r]; // already * -log2e
            if (t < 8) {
                #pragma unroll
                for (int j = 0; j < 8; ++j) {
                    const float tt = fmaf(NEG_LOG2E, acc[j], b1);
                    sacc[j] += w2 * silu_t(tt);
                }
            } else {
                #pragma unroll
                for (int j = 0; j < 8; ++j) {
                    const float h = acc[j];
                    #pragma unroll
                    for (int m = 0; m < 5; ++m) {
                        const float tt = fmaf(sphs[j][m], h, b1);
                        iacc[j][m] += w2 * silu_t(tt);
                    }
                }
            }
        }

        // ---- reduce partial channel-sums across the 16 lanes of each half-wave
        #pragma unroll
        for (int j = 0; j < 8; ++j) {
            float s = sacc[j];
            for (int off = 1; off < 16; off <<= 1) s += __shfl_xor(s, off, 16);
            sacc[j] = s + c_bs2;
            #pragma unroll
            for (int m = 0; m < 5; ++m) {
                float v = iacc[j][m];
                for (int off = 1; off < 16; off <<= 1) v += __shfl_xor(v, off, 16);
                iacc[j][m] = v + c_bi2;
            }
        }

        // ---- segmented sum into node accumulators (one lane per edge)
        if (n16 == 0) {
            #pragma unroll
            for (int j = 0; j < 8; ++j) {
                const int e = base + j + 8 * g;
                if (e < E) {
                    const int node = edge_index[e];
                    atomicAdd(&nodeScalar[node], sacc[j]);
                    atomicAdd(&nodeCnt[node], 1.0f);
                    #pragma unroll
                    for (int m = 0; m < 5; ++m)
                        atomicAdd(&nodeIrrep[node * 5 + m], iacc[j][m]);
                }
            }
        }
    }
}

// ---------------------------------------------------------------------------
__global__ void zero_kernel(float* __restrict__ p, int n) {
    const int i = blockIdx.x * blockDim.x + threadIdx.x;
    if (i < n) p[i] = 0.0f;
}

// node means -> per-graph accumulators (LDS pre-reduction; batch is sorted)
__global__ __launch_bounds__(256) void node_kernel(
    const float* __restrict__ nodeScalar, const float* __restrict__ nodeIrrep,
    const float* __restrict__ nodeCnt, const int* __restrict__ batch,
    int N,
    float* __restrict__ gScalar, float* __restrict__ gIrrep, float* __restrict__ gCnt)
{
    __shared__ float ls[16], li[16 * 5], lc[16];
    const int tid = threadIdx.x;
    if (tid < 16) { ls[tid] = 0.0f; lc[tid] = 0.0f; }
    if (tid < 80) li[tid] = 0.0f;
    __syncthreads();

    const int i = blockIdx.x * blockDim.x + tid;
    if (i < N) {
        const float inv = 1.0f / fmaxf(nodeCnt[i], 1.0f);
        const int b = batch[i];
        if (b >= 0 && b < 16) {
            atomicAdd(&ls[b], nodeScalar[i] * inv);
            atomicAdd(&lc[b], 1.0f);
            #pragma unroll
            for (int m = 0; m < 5; ++m)
                atomicAdd(&li[b * 5 + m], nodeIrrep[i * 5 + m] * inv);
        }
    }
    __syncthreads();
    if (tid < 16) { atomicAdd(&gScalar[tid], ls[tid]); atomicAdd(&gCnt[tid], lc[tid]); }
    if (tid < 80) atomicAdd(&gIrrep[tid], li[tid]);
}

__global__ void final_kernel(const float* __restrict__ gScalar,
                             const float* __restrict__ gIrrep,
                             const float* __restrict__ gCnt,
                             float* __restrict__ out, int B)
{
    const int tid = blockIdx.x * blockDim.x + threadIdx.x;
    if (tid < B) {
        out[tid] = gScalar[tid] / fmaxf(gCnt[tid], 1.0f);
    } else if (tid < B + 5 * B) {
        const int idx = tid - B;
        const int b = idx / 5;
        out[tid] = gIrrep[idx] / fmaxf(gCnt[b], 1.0f);
    }
}

// ---------------------------------------------------------------------------
extern "C" void kernel_launch(void* const* d_in, const int* in_sizes, int n_in,
                              void* d_out, int out_size, void* d_ws, size_t ws_size,
                              hipStream_t stream)
{
    (void)n_in; (void)out_size; (void)ws_size;
    const float* x_edge     = (const float*)d_in[0];
    const float* edge_vec   = (const float*)d_in[1];
    const int*   edge_index = (const int*)d_in[2];
    const int*   batch      = (const int*)d_in[3];
    // d_in[4] = natoms: only its length (B) matters
    const float* Ws1 = (const float*)d_in[5];
    const float* bs1 = (const float*)d_in[6];
    const float* Ws2 = (const float*)d_in[7];
    const float* bs2 = (const float*)d_in[8];
    const float* Wi1 = (const float*)d_in[9];
    const float* bi1 = (const float*)d_in[10];
    const float* Wi2 = (const float*)d_in[11];
    const float* bi2 = (const float*)d_in[12];

    const int E = in_sizes[0] / EMB;
    const int N = in_sizes[3];
    const int B = in_sizes[4];

    float* ws         = (float*)d_ws;
    float* nodeScalar = ws;                    // N
    float* nodeIrrep  = nodeScalar + N;        // 5N
    float* nodeCnt    = nodeIrrep + 5 * N;     // N
    float* gScalar    = nodeCnt + N;           // B
    float* gIrrep     = gScalar + B;           // 5B
    float* gCnt       = gIrrep + 5 * B;        // B

    const int ztotal = 7 * N + 7 * B;
    zero_kernel<<<(ztotal + 255) / 256, 256, 0, stream>>>(ws, ztotal);

    const int blocks = 256; // 2048 waves, grid-stride over E/16 = 20000 tiles
    edge_kernel<<<blocks, 256, 0, stream>>>(
        x_edge, edge_vec, edge_index,
        Ws1, bs1, Ws2, bs2, Wi1, bi1, Wi2, bi2,
        nodeScalar, nodeIrrep, nodeCnt, E);

    node_kernel<<<(N + 255) / 256, 256, 0, stream>>>(
        nodeScalar, nodeIrrep, nodeCnt, batch, N, gScalar, gIrrep, gCnt);

    final_kernel<<<1, 256, 0, stream>>>(gScalar, gIrrep, gCnt, (float*)d_out, B);
}
